// Net_74723841016020
// MI455X (gfx1250) — matrix operand order repaired and checked by reading, not compiled, not run on hardware
//
#include <hip/hip_runtime.h>
#include <hip/hip_bf16.h>

typedef __attribute__((ext_vector_type(16))) _Float16 v16h;
typedef __attribute__((ext_vector_type(8)))  float    v8f;

// Problem constants (fixed by reference setup_inputs)
#define BATCH 4
#define Himg  128
#define Wimg  128
#define HW    16384          // 128*128
#define NPIX  65536          // BATCH*HW

// ---------------------------------------------------------------------------
// Weight pre-swizzle: pack OIHW f32 weights into WMMA B-fragment layout f16.
// Fragment layout: flat[( (t*KSTEPS + s)*32 + lane )*16 + j]
//   h = lane>>4, m = lane&15, k = s*32 + 16*h + j, co = t*16 + m
//   k-major ordering k = n*CIN + ci  (n = 3x3 tap index a*3+b)
// Each lane's 16-half fragment is 32 contiguous bytes -> two b128 loads.
// ---------------------------------------------------------------------------
__device__ inline void write_frag(_Float16* __restrict__ dst,
                                  const float* __restrict__ W,
                                  int i, int CIN, int CoValid, int KSTEPS) {
    int j    = i & 15;
    int lane = (i >> 4) & 31;
    int s    = (i >> 9) % KSTEPS;
    int t    = (i >> 9) / KSTEPS;
    int h = lane >> 4, m = lane & 15;
    int k  = s * 32 + 16 * h + j;
    int co = t * 16 + m;
    int ci = k % CIN, n = k / CIN;
    float v = (co < CoValid) ? W[(co * CIN + ci) * 9 + n] : 0.0f;
    dst[i] = (_Float16)v;
}

__global__ void prep_weights(const float* __restrict__ wo2, const float* __restrict__ w2,
                             const float* __restrict__ wo3, const float* __restrict__ w3,
                             _Float16* __restrict__ o2h, _Float16* __restrict__ w2h,
                             _Float16* __restrict__ o3h, _Float16* __restrict__ w3h) {
    int i = blockIdx.x * blockDim.x + threadIdx.x;
    const int S_O2 = 2 * 18 * 512;   // 18432: NT=2, K=576 (CIN 64), valid 18
    const int S_W2 = 2 * 18 * 512;   // 18432: NT=2, K=576,          valid 32
    const int S_O3 = 2 * 9 * 512;    //  9216: NT=2, K=288 (CIN 32), valid 18
    const int S_W3 = 1 * 9 * 512;    //  4608: NT=1, K=288,          valid 4
    if (i < S_O2)                               write_frag(o2h, wo2, i,                     64, 18, 18);
    else if (i < S_O2 + S_W2)                   write_frag(w2h, w2,  i - S_O2,              64, 32, 18);
    else if (i < S_O2 + S_W2 + S_O3)            write_frag(o3h, wo3, i - S_O2 - S_W2,       32, 18, 9);
    else if (i < S_O2 + S_W2 + S_O3 + S_W3)     write_frag(w3h, w3,  i - S_O2 - S_W2 - S_O3,32,  4, 9);
}

// ---------------------------------------------------------------------------
// Shared WMMA GEMM inner loop: C[16 x NT*16] += A[16 x K] * B[K x NT*16].
// A rows live in LDS (f16, 16B-aligned, stride K+8), B fragments are
// pre-swizzled in global memory (contiguous 32B per lane per fragment).
// ---------------------------------------------------------------------------
template <int K, int NT>
__device__ inline void wmma_gemm(const _Float16 (*__restrict__ S)[K + 8],
                                 const _Float16* __restrict__ wB,
                                 int lane, v8f* acc) {
    constexpr int KSTEPS = K / 32;
    const int h = lane >> 4;
    const int m = lane & 15;
#pragma unroll
    for (int s = 0; s < KSTEPS; ++s) {
        // A fragment: lane half h holds K = s*32+16h+0..15 (elems 0-15)
        // -> two 16-byte DS loads.
        union { v16h v; uint4 q[2]; } af;
        const uint4* rp = reinterpret_cast<const uint4*>(&S[m][s * 32 + 16 * h]);
        af.q[0] = rp[0];
        af.q[1] = rp[1];
#pragma unroll
        for (int t = 0; t < NT; ++t) {
            union { v16h v; uint4 q[2]; } bf;
            const uint4* bp = reinterpret_cast<const uint4*>(wB)
                            + ((size_t)((t * KSTEPS + s) * 32 + lane)) * 2;
            bf.q[0] = bp[0];
            bf.q[1] = bp[1];
            acc[t] = __builtin_amdgcn_wmma_f32_16x16x32_f16(
                false, af.v, false, bf.v, (short)0, acc[t], false, false);
        }
    }
}

// ---------------------------------------------------------------------------
// Kernel 1: conv1 1->64, 3x3, pad 1, relu.  One thread per pixel, Cin==1 so
// the 9-tap neighborhood is loaded once and reused for all 64 out channels.
// ---------------------------------------------------------------------------
__global__ void conv1_relu(const float* __restrict__ x,
                           const float* __restrict__ w1,
                           const float* __restrict__ b1,
                           float* __restrict__ h1) {
    int idx = blockIdx.x * blockDim.x + threadIdx.x;   // NPIX threads
    int b = idx >> 14;
    int p = idx & (HW - 1);
    int y = p >> 7, xx = p & 127;
    const float* xb = x + ((size_t)b << 14);
    float v[9];
#pragma unroll
    for (int ky = 0; ky < 3; ++ky)
#pragma unroll
        for (int kx = 0; kx < 3; ++kx) {
            int yy = y + ky - 1, xc = xx + kx - 1;
            bool ok = (yy >= 0) & (yy < Himg) & (xc >= 0) & (xc < Wimg);
            v[ky * 3 + kx] = ok ? xb[yy * Wimg + xc] : 0.0f;
        }
    for (int co = 0; co < 64; ++co) {
        float acc = b1[co];
#pragma unroll
        for (int t = 0; t < 9; ++t) acc += w1[co * 9 + t] * v[t];
        h1[(((size_t)(b * 64 + co)) << 14) + p] = fmaxf(acc, 0.0f);
    }
}

// ---------------------------------------------------------------------------
// Kernels 2 & 4: 3x3 pad-1 conv Cin -> 18 via WMMA (N padded to 32 cols with
// zero weights).  One wave per 16-pixel tile; im2col A-matrix staged in LDS.
// ---------------------------------------------------------------------------
template <int CIN>
__global__ void __launch_bounds__(32)
conv3x3_wmma(const float* __restrict__ hin,
             const _Float16* __restrict__ wB,
             const float* __restrict__ bias,
             float* __restrict__ out) {
    constexpr int K      = CIN * 9;
    constexpr int STRIDE = K + 8;
    constexpr int COV    = 18;

    __shared__ alignas(16) _Float16 S[16][STRIDE];

    const int lane     = threadIdx.x;
    const int tileBase = blockIdx.x * 16;
    const int h        = lane >> 4;
    const int m        = lane & 15;

    // ---- im2col: S[p][n*CIN+ci] = padded 3x3 neighborhood ----------------
    for (int j = lane; j < 16 * 9 * CIN; j += 32) {
        int p    = j & 15;
        int rest = j >> 4;
        int ci   = rest % CIN;
        int n    = rest / CIN;
        int pg   = tileBase + p;
        int b    = pg >> 14;
        int pp   = pg & (HW - 1);
        int y = pp >> 7, x = pp & 127;
        int yy = y + n / 3 - 1, xc = x + n % 3 - 1;
        float v = (yy >= 0 && yy < Himg && xc >= 0 && xc < Wimg)
                ? hin[(((size_t)(b * CIN + ci)) << 14) + yy * Wimg + xc] : 0.0f;
        S[p][n * CIN + ci] = (_Float16)v;
    }
    __syncthreads();

    v8f acc[2];
#pragma unroll
    for (int t = 0; t < 2; ++t) {
        int co = t * 16 + m;
        float bv = (co < COV) ? bias[co] : 0.0f;
#pragma unroll
        for (int r = 0; r < 8; ++r) acc[t][r] = bv;
    }

    wmma_gemm<K, 2>(S, wB, lane, acc);

#pragma unroll
    for (int t = 0; t < 2; ++t)
#pragma unroll
        for (int r = 0; r < 8; ++r) {
            int co = t * 16 + m;
            if (co < COV) {
                int pg = tileBase + r + 8 * h;
                int b  = pg >> 14;
                int p  = pg & (HW - 1);
                out[(((size_t)(b * 18 + co)) << 14) + p] = acc[t][r];
            }
        }
}

// ---------------------------------------------------------------------------
// Kernels 3 & 5: fused deform_sample + stride-3 conv == per-pixel GEMM.
// FINAL=true additionally applies pixel_shuffle(2) on store.
// ---------------------------------------------------------------------------
template <int CIN, int NT, bool FINAL>
__global__ void __launch_bounds__(32)
deform_conv_wmma(const float* __restrict__ hin,
                 const float* __restrict__ off,
                 const _Float16* __restrict__ wB,
                 const float* __restrict__ bias,
                 float* __restrict__ out) {
    constexpr int K      = CIN * 9;
    constexpr int STRIDE = K + 8;
    constexpr int NC     = NT * 16;

    __shared__ alignas(16) _Float16 S[16][STRIDE];
    __shared__ int   O4[144][4];
    __shared__ float Wt[144][4];

    const int lane     = threadIdx.x;
    const int tileBase = blockIdx.x * 16;
    const int h        = lane >> 4;
    const int m        = lane & 15;

    // ---- Phase 0: per-(pixel, tap) bilinear sampling coordinates ---------
    for (int e = lane; e < 144; e += 32) {
        int p  = e / 9, n = e % 9;
        int pg = tileBase + p;
        int b  = pg >> 14;
        int pp = pg & (HW - 1);
        int y = pp >> 7, x = pp & 127;
        float oy = off[(((size_t)(b * 18 + n)) << 14) + pp];
        float ox = off[(((size_t)(b * 18 + 9 + n)) << 14) + pp];
        float py = oy + (float)(n / 3 - 1) + (float)y;
        float px = ox + (float)(n % 3 - 1) + (float)x;
        py = fminf(fmaxf(py, 0.0f), 127.0f);
        px = fminf(fmaxf(px, 0.0f), 127.0f);
        float fy = floorf(py), fx = floorf(px);
        int y0 = (int)fy, x0 = (int)fx;
        int y1 = min(y0 + 1, 127), x1 = min(x0 + 1, 127);
        float wy = py - fy, wx = px - fx;
        O4[e][0] = y0 * Wimg + x0;  O4[e][1] = y0 * Wimg + x1;
        O4[e][2] = y1 * Wimg + x0;  O4[e][3] = y1 * Wimg + x1;
        Wt[e][0] = (1.0f - wy) * (1.0f - wx);
        Wt[e][1] = (1.0f - wy) * wx;
        Wt[e][2] = wy * (1.0f - wx);
        Wt[e][3] = wy * wx;
    }
    __syncthreads();

    // ---- Phase 1: bilinear gather -> f16 A-matrix in LDS (k = n*CIN+ci) --
    for (int j = lane; j < 16 * 9 * CIN; j += 32) {
        int p    = j & 15;
        int rest = j >> 4;
        int ci   = rest % CIN;
        int n    = rest / CIN;
        int e    = p * 9 + n;
        int pg   = tileBase + p;
        int b    = pg >> 14;
        const float* base = hin + (((size_t)(b * CIN + ci)) << 14);
        float v = Wt[e][0] * base[O4[e][0]] + Wt[e][1] * base[O4[e][1]]
                + Wt[e][2] * base[O4[e][2]] + Wt[e][3] * base[O4[e][3]];
        S[p][n * CIN + ci] = (_Float16)v;
    }
    __syncthreads();

    // ---- Phase 2: WMMA GEMM, bias folded into accumulator init -----------
    v8f acc[NT];
#pragma unroll
    for (int t = 0; t < NT; ++t) {
        float bv = FINAL ? ((m < 4) ? bias[m] : 0.0f) : bias[t * 16 + m];
#pragma unroll
        for (int r = 0; r < 8; ++r) acc[t][r] = bv;
    }

    wmma_gemm<K, NT>(S, wB, lane, acc);

    // ---- Phase 3: store (relu for mid layer, pixel_shuffle for final) ----
#pragma unroll
    for (int t = 0; t < NT; ++t)
#pragma unroll
        for (int r = 0; r < 8; ++r) {
            int pg = tileBase + r + 8 * h;
            int b  = pg >> 14;
            int p  = pg & (HW - 1);
            int co = t * 16 + m;
            float v = acc[t][r];
            if (!FINAL) {
                out[(((size_t)(b * NC + co)) << 14) + p] = fmaxf(v, 0.0f);
            } else if (co < 4) {
                int y = p >> 7, x = p & 127;
                int a = co >> 1, c = co & 1;
                out[(size_t)b * 65536 + (2 * y + a) * 256 + (2 * x + c)] = v;
            }
        }
}

// ---------------------------------------------------------------------------
extern "C" void kernel_launch(void* const* d_in, const int* in_sizes, int n_in,
                              void* d_out, int out_size, void* d_ws, size_t ws_size,
                              hipStream_t stream) {
    const float* x   = (const float*)d_in[0];
    const float* w1  = (const float*)d_in[1];
    const float* b1  = (const float*)d_in[2];
    const float* wo2 = (const float*)d_in[3];
    const float* bo2 = (const float*)d_in[4];
    const float* w2  = (const float*)d_in[5];
    const float* b2  = (const float*)d_in[6];
    const float* wo3 = (const float*)d_in[7];
    const float* bo3 = (const float*)d_in[8];
    const float* w3  = (const float*)d_in[9];
    const float* b3  = (const float*)d_in[10];

    // Workspace carve-out (floats): h1 | off2 | h2 | off3 | f16 frag weights
    float* h1   = (float*)d_ws;                 // 4*64*HW  = 4,194,304
    float* off2 = h1 + (size_t)4 * 64 * HW;     // 4*18*HW  = 1,179,648
    float* h2   = off2 + (size_t)4 * 18 * HW;   // 4*32*HW  = 2,097,152
    float* off3 = h2 + (size_t)4 * 32 * HW;     // 4*18*HW  = 1,179,648
    _Float16* o2h = (_Float16*)(off3 + (size_t)4 * 18 * HW);
    _Float16* w2h = o2h + 2 * 18 * 512;         // 18432 halves each
    _Float16* o3h = w2h + 2 * 18 * 512;
    _Float16* w3h = o3h + 2 * 9 * 512;

    const int NFRAG = 2 * 18 * 512 + 2 * 18 * 512 + 2 * 9 * 512 + 1 * 9 * 512; // 50688
    prep_weights<<<(NFRAG + 255) / 256, 256, 0, stream>>>(wo2, w2, wo3, w3,
                                                          o2h, w2h, o3h, w3h);
    conv1_relu<<<NPIX / 256, 256, 0, stream>>>(x, w1, b1, h1);
    conv3x3_wmma<64><<<NPIX / 16, 32, 0, stream>>>(h1, o2h, bo2, off2);
    deform_conv_wmma<64, 2, false><<<NPIX / 16, 32, 0, stream>>>(h1, off2, w2h, b2, h2);
    conv3x3_wmma<32><<<NPIX / 16, 32, 0, stream>>>(h2, o3h, bo3, off3);
    deform_conv_wmma<32, 1, true><<<NPIX / 16, 32, 0, stream>>>(h2, off3, w3h, b3, (float*)d_out);
}
